// SCPositionwiseFeedForward_low_15659450761494
// MI455X (gfx1250) — matrix operand order repaired and checked by reading, not compiled, not run on hardware
//
#include <hip/hip_runtime.h>
#include <hip/hip_bf16.h>
#include <stdint.h>

typedef unsigned short u16;
typedef unsigned int   u32;
typedef __attribute__((ext_vector_type(16))) __bf16 v16bf_t;
typedef __attribute__((ext_vector_type(8)))  float  v8f_t;

#define B_   16
#define T_   1024
#define DIN  384
#define H1_  1024
#define H2_  1024
#define K_   9

// ---------- bf16 helpers (raw u16 storage, RNE convert) ----------
__device__ __forceinline__ u16 f2bf(float f) {
  union { float f; u32 u; } v; v.f = f;
  u32 r = v.u + 0x7FFFu + ((v.u >> 16) & 1u);
  return (u16)(r >> 16);
}
__device__ __forceinline__ float bf2f(u16 h) {
  union { u32 u; float f; } v; v.u = ((u32)h) << 16; return v.f;
}

// ---------- CDNA5 async memory->LDS copy (ASYNCcnt-tracked) ----------
__device__ __forceinline__ void async_load_b128(u32 lds_off, const u16* gptr) {
  asm volatile("global_load_async_to_lds_b128 %0, %1, off"
               :: "v"(lds_off), "v"(gptr) : "memory");
}
__device__ __forceinline__ void wait_async0() {
  asm volatile("s_wait_asynccnt 0x0" ::: "memory");
}

// ---------- elementwise f32 -> bf16 cast ----------
__global__ void cvt_f32_bf16(const float* __restrict__ src, u16* __restrict__ dst, long n) {
  long i = (long)blockIdx.x * blockDim.x + threadIdx.x;
  if (i < n) dst[i] = f2bf(src[i]);
}

// ---------- inv L2 norm of d_w over H1, per (b,k) ----------
__global__ void norm_d_kernel(const float* __restrict__ d_w, float* __restrict__ inv_d) {
  __shared__ float red[256];
  int b = blockIdx.x / K_;
  int k = blockIdx.x % K_;
  float s = 0.f;
  for (int h = threadIdx.x; h < H1_; h += 256) {
    float v = d_w[((long)b * H1_ + h) * K_ + k];
    s += v * v;
  }
  red[threadIdx.x] = s;
  __syncthreads();
  for (int off = 128; off > 0; off >>= 1) {
    if (threadIdx.x < off) red[threadIdx.x] += red[threadIdx.x + off];
    __syncthreads();
  }
  if (threadIdx.x == 0)
    inv_d[b * K_ + k] = 1.f / fmaxf(sqrtf(red[0]), 1e-12f);
}

// ---------- dwn[b][c][k] = d_w * inv_d[b][k] * d_g[b][c] ----------
__global__ void dwn_kernel(const float* __restrict__ d_w, const float* __restrict__ d_g,
                           const float* __restrict__ inv_d, float* __restrict__ dwn) {
  int i = blockIdx.x * 256 + threadIdx.x;       // exactly B*H1*K threads
  int k = i % K_;
  int c = (i / K_) % H1_;
  int b = i / (K_ * H1_);
  dwn[i] = d_w[i] * inv_d[b * K_ + k] * d_g[b * H1_ + c];
}

// ---------- inv L2 norm of p_w over H1, per (b,o) ----------
__global__ void norm_p_kernel(const float* __restrict__ p_w, float* __restrict__ inv_p) {
  __shared__ float red[256];
  int b  = blockIdx.x >> 4;
  int o  = ((blockIdx.x & 15) << 6) + (threadIdx.x & 63);
  int cg = threadIdx.x >> 6;                    // 0..3
  float s = 0.f;
  for (int c = cg; c < H1_; c += 4) {
    float v = p_w[((long)b * H1_ + c) * H2_ + o];
    s += v * v;
  }
  red[threadIdx.x] = s;
  __syncthreads();
  if (cg == 0) {
    float tot = red[threadIdx.x] + red[threadIdx.x + 64] +
                red[threadIdx.x + 128] + red[threadIdx.x + 192];
    inv_p[b * H2_ + o] = 1.f / fmaxf(sqrtf(tot), 1e-12f);
  }
}

// ---------- pn_t[b][o][c] = p_w[b][c][o] * inv_p[b][o] * p_g[b][c] (bf16), tiled transpose ----------
__global__ void pn_transpose_kernel(const float* __restrict__ p_w, const float* __restrict__ p_g,
                                    const float* __restrict__ inv_p, u16* __restrict__ pn_t) {
  __shared__ float tile[32][33];
  int b  = blockIdx.z;
  int c0 = blockIdx.x * 32, o0 = blockIdx.y * 32;
  int tx = threadIdx.x & 31, ty = threadIdx.x >> 5;   // 32 x 8
#pragma unroll
  for (int i = 0; i < 4; ++i) {
    int c = c0 + ty + i * 8;
    tile[ty + i * 8][tx] = p_w[((long)b * H1_ + c) * H2_ + o0 + tx];
  }
  __syncthreads();
  float g = p_g[b * H1_ + c0 + tx];
#pragma unroll
  for (int i = 0; i < 4; ++i) {
    int o = o0 + ty + i * 8;
    float v = tile[tx][ty + i * 8] * inv_p[b * H2_ + o] * g;
    pn_t[((long)b * H2_ + o) * H1_ + c0 + tx] = f2bf(v);
  }
}

// ---------- depthwise conv along T, (B,T,H1) layout, y = conv*T + d_b ----------
__global__ void dwconv_kernel(const u16* __restrict__ x, const float* __restrict__ dwn,
                              const float* __restrict__ d_b, u16* __restrict__ y) {
  int i = blockIdx.x * 256 + threadIdx.x;   // B*T*H1 threads
  int c = i & (H1_ - 1);
  int t = (i >> 10) & (T_ - 1);
  int b = i >> 20;
  const float* w = dwn + ((long)b * H1_ + c) * K_;
  float acc = 0.f;
#pragma unroll
  for (int k = 0; k < K_; ++k) {
    int tt = t + k - 4;
    if (tt >= 0 && tt < T_)
      acc += bf2f(x[((long)b * T_ + tt) * H1_ + c]) * w[k];
  }
  acc = acc * (float)T_ + d_b[b * H1_ + c];
  y[((long)b * T_ + t) * H1_ + c] = f2bf(acc);
}

// ---------- bf16 WMMA GEMM: C(MxN) = A(MxK) * B(KxN), A row-major, B stored [N][K] ----------
// macro tile 128x128, 8 waves (4 M x 2 N), wave tile 32x64 (2x4 wmma), K-step 32,
// double-buffered LDS filled by global_load_async_to_lds_b128 (one barrier per K-step)
#define TM 128
#define TN 128
#define TK 32
#define LDS_STRIDE 34   // +2 pad kills bank conflicts (17 uints/row, coprime with 64 banks)

union FragU { u32 u[8]; v16bf_t v; };

template <int EPI>
__global__ __launch_bounds__(256) void gemm_bf16_wmma(
    const u16* __restrict__ A,  long strideA, int lda,
    const u16* __restrict__ Bm, long strideB, int ldb,
    int Kdim,
    const float* __restrict__ bias, int biasStride,
    const float* __restrict__ resid,         // EPI==3 only (same layout as out)
    u16* __restrict__ outBf,                 // EPI==1,2
    long strideOut, int ldo,
    float* __restrict__ outF)                // EPI==3
{
  __shared__ u16 sA[2][TM * LDS_STRIDE];
  __shared__ u16 sB[2][TN * LDS_STRIDE];

  const int tid  = threadIdx.x;
  const int m0   = blockIdx.x * TM;
  const int n0   = blockIdx.y * TN;
  const u16* Ab  = A  + (long)blockIdx.z * strideA + (long)m0 * lda;
  const u16* Bb  = Bm + (long)blockIdx.z * strideB + (long)n0 * ldb;

  // cooperative tile-fill coords: each thread copies 16 bf16 (2 x b128) of A and of B
  const int trow = tid >> 1;            // 0..127
  const int tk   = (tid & 1) * 16;      // 0 or 16

  const int wid  = tid >> 5;
  const int lane = tid & 31;
  const int lo   = lane & 15;
  const int hi   = lane >> 4;
  const int wm   = (wid >> 1) * 32;     // wave M offset (4 waves)
  const int wn   = (wid & 1) * 64;      // wave N offset (2 waves)

  v8f_t acc[2][4] = {};

  // prologue: fill buffer 0
  {
    u32 aoff = (u32)(uintptr_t)(&sA[0][trow * LDS_STRIDE + tk]);
    u32 boff = (u32)(uintptr_t)(&sB[0][trow * LDS_STRIDE + tk]);
    const u16* ap = Ab + (long)trow * lda + tk;
    const u16* bp = Bb + (long)trow * ldb + tk;
    async_load_b128(aoff,      ap);
    async_load_b128(aoff + 16, ap + 8);
    async_load_b128(boff,      bp);
    async_load_b128(boff + 16, bp + 8);
  }

  int cur = 0;
  for (int k0 = 0; k0 < Kdim; k0 += TK, cur ^= 1) {
    wait_async0();        // this wave's async fills (and prior iter's) are in LDS
    __syncthreads();      // all waves filled buf[cur]; all done reading buf[cur^1]

    if (k0 + TK < Kdim) { // kick next tile into the other buffer
      int nxt = cur ^ 1;
      u32 aoff = (u32)(uintptr_t)(&sA[nxt][trow * LDS_STRIDE + tk]);
      u32 boff = (u32)(uintptr_t)(&sB[nxt][trow * LDS_STRIDE + tk]);
      const u16* ap = Ab + (long)trow * lda + k0 + TK + tk;
      const u16* bp = Bb + (long)trow * ldb + k0 + TK + tk;
      async_load_b128(aoff,      ap);
      async_load_b128(aoff + 16, ap + 8);
      async_load_b128(boff,      bp);
      async_load_b128(boff + 16, bp + 8);
    }

    const u32* sAu = reinterpret_cast<const u32*>(sA[cur]);
    const u32* sBu = reinterpret_cast<const u32*>(sB[cur]);

    // A fragments (16x32, ISA 7.12.2: K = 2*(v&3) + 16*(v>>2) + 8*hi)
    FragU a[2], b[4];
#pragma unroll
    for (int mi = 0; mi < 2; ++mi) {
      int base = (wm + mi * 16 + lo) * 17 + hi * 4;
#pragma unroll
      for (int v = 0; v < 8; ++v)
        a[mi].u[v] = sAu[base + (v & 3) + (v >> 2) * 8];
    }
  // B fragments (32x16: N = lane&15, K = 2*(v&3) + 16*(v>>2) + 8*hi)
#pragma unroll
  for (int ni = 0; ni < 4; ++ni) {
      int base = (wn + ni * 16 + lo) * 17 + hi * 4;
#pragma unroll
      for (int v = 0; v < 8; ++v)
        b[ni].u[v] = sBu[base + (v & 3) + (v >> 2) * 8];
    }
#pragma unroll
    for (int mi = 0; mi < 2; ++mi)
#pragma unroll
      for (int ni = 0; ni < 4; ++ni)
        acc[mi][ni] = __builtin_amdgcn_wmma_f32_16x16x32_bf16(
            false, a[mi].v, false, b[ni].v, (short)0, acc[mi][ni], false, false);
  }

  // epilogue: C/D layout: row = v + 8*hi, col = lane&15
  const long outOff = (long)blockIdx.z * strideOut;
#pragma unroll
  for (int mi = 0; mi < 2; ++mi)
#pragma unroll
    for (int ni = 0; ni < 4; ++ni)
#pragma unroll
      for (int v = 0; v < 8; ++v) {
        int row = m0 + wm + mi * 16 + hi * 8 + v;
        int col = n0 + wn + ni * 16 + lo;
        float val = acc[mi][ni][v] + bias[(long)blockIdx.z * biasStride + col];
        if constexpr (EPI == 1) {        // mish
          float sp = val > 15.f ? val : log1pf(__expf(val));
          val = val * tanhf(sp);
          outBf[outOff + (long)row * ldo + col] = f2bf(val);
        } else if constexpr (EPI == 2) { // bias only
          outBf[outOff + (long)row * ldo + col] = f2bf(val);
        } else {                         // bias + residual, f32 out
          long idx = outOff + (long)row * ldo + col;
          outF[idx] = val + resid[idx];
        }
      }
}

extern "C" void kernel_launch(void* const* d_in, const int* in_sizes, int n_in,
                              void* d_out, int out_size, void* d_ws, size_t ws_size,
                              hipStream_t stream) {
  const float* input = (const float*)d_in[0];
  const float* d_w   = (const float*)d_in[1];
  const float* d_g   = (const float*)d_in[2];
  const float* d_b   = (const float*)d_in[3];
  const float* p_w   = (const float*)d_in[4];
  const float* p_g   = (const float*)d_in[5];
  const float* p_b   = (const float*)d_in[6];
  const float* w1_w  = (const float*)d_in[7];
  const float* w1_b  = (const float*)d_in[8];
  const float* w2_w  = (const float*)d_in[9];
  const float* w2_b  = (const float*)d_in[10];
  float* out = (float*)d_out;

  char* ws = (char*)d_ws;
  size_t off = 0;
  auto alloc = [&](size_t bytes) -> void* {
    void* p = ws + off;
    off += (bytes + 255) & ~(size_t)255;
    return p;
  };
  u16*   in_bf  = (u16*)  alloc((size_t)B_ * T_ * DIN * 2);
  u16*   w1_bf  = (u16*)  alloc((size_t)H1_ * DIN * 2);
  u16*   w2_bf  = (u16*)  alloc((size_t)DIN * H2_ * 2);
  u16*   x_bf   = (u16*)  alloc((size_t)B_ * T_ * H1_ * 2);
  u16*   y_bf   = (u16*)  alloc((size_t)B_ * T_ * H1_ * 2);
  u16*   z_bf   = (u16*)  alloc((size_t)B_ * T_ * H2_ * 2);
  u16*   pn_t   = (u16*)  alloc((size_t)B_ * H2_ * H1_ * 2);
  float* inv_d  = (float*)alloc((size_t)B_ * K_ * 4);
  float* inv_p  = (float*)alloc((size_t)B_ * H2_ * 4);
  float* dwn_ws = (float*)alloc((size_t)B_ * H1_ * K_ * 4);

  // prep
  cvt_f32_bf16<<<(B_ * T_ * DIN + 255) / 256, 256, 0, stream>>>(input, in_bf, (long)B_ * T_ * DIN);
  cvt_f32_bf16<<<(H1_ * DIN + 255) / 256, 256, 0, stream>>>(w1_w, w1_bf, (long)H1_ * DIN);
  cvt_f32_bf16<<<(DIN * H2_ + 255) / 256, 256, 0, stream>>>(w2_w, w2_bf, (long)DIN * H2_);
  norm_d_kernel<<<B_ * K_, 256, 0, stream>>>(d_w, inv_d);
  dwn_kernel<<<(B_ * H1_ * K_) / 256, 256, 0, stream>>>(d_w, d_g, inv_d, dwn_ws);
  norm_p_kernel<<<B_ * (H2_ / 64), 256, 0, stream>>>(p_w, inv_p);
  pn_transpose_kernel<<<dim3(H1_ / 32, H2_ / 32, B_), 256, 0, stream>>>(p_w, p_g, inv_p, pn_t);

  // GEMM1: h = mish(input @ w1^T + b1)  -> x_bf (B,T,H1)
  gemm_bf16_wmma<1><<<dim3(T_ / TM, H1_ / TN, B_), 256, 0, stream>>>(
      in_bf, (long)T_ * DIN, DIN, w1_bf, 0, DIN, DIN,
      w1_b, 0, nullptr, x_bf, (long)T_ * H1_, H1_, nullptr);

  // depthwise conv -> y_bf (B,T,H1)
  dwconv_kernel<<<(B_ * T_ * H1_) / 256, 256, 0, stream>>>(x_bf, dwn_ws, d_b, y_bf);

  // GEMM2 (pointwise): z = y @ pn_t^T + p_b -> z_bf (B,T,H2)
  gemm_bf16_wmma<2><<<dim3(T_ / TM, H2_ / TN, B_), 256, 0, stream>>>(
      y_bf, (long)T_ * H1_, H1_, pn_t, (long)H2_ * H1_, H1_, H1_,
      p_b, H2_, nullptr, z_bf, (long)T_ * H2_, H2_, nullptr);

  // GEMM3: out = z @ w2^T + b2 + residual -> f32 out (B,T,DIN)
  gemm_bf16_wmma<3><<<dim3(T_ / TM, DIN / TN, B_), 256, 0, stream>>>(
      z_bf, (long)T_ * H2_, H2_, w2_bf, 0, H2_, H2_,
      w2_b, 0, input, nullptr, (long)T_ * DIN, DIN, out);
}
